// AttentionHead_1803886265630
// MI455X (gfx1250) — matrix operand order repaired and checked by reading, not compiled, not run on hardware
//
#include <hip/hip_runtime.h>
#include <hip/hip_bf16.h>

// Problem constants (from reference): B=8, T=4096, E=768, H=64
#define BT_TOTAL 32768   // B*T
#define TSEQ     4096
#define EMB      768
#define HS       64
#define KSTEPS   (EMB / 32)   // 24

typedef __attribute__((ext_vector_type(16))) __bf16 v16bf;
typedef __attribute__((ext_vector_type(8)))  __bf16 v8bf;
typedef __attribute__((ext_vector_type(8)))  float  v8f;
typedef __attribute__((ext_vector_type(4)))  unsigned int v4u;
typedef __attribute__((ext_vector_type(8)))  int v8i;
typedef __attribute__((ext_vector_type(4)))  int v4i;

static __device__ __forceinline__ v16bf join16(v8bf lo, v8bf hi) {
    return __builtin_shufflevector(lo, hi, 0,1,2,3,4,5,6,7,8,9,10,11,12,13,14,15);
}

// ---------------------------------------------------------------------------
// TDM 2D tile load: global (row-major, elem stride0 between rows) -> LDS
// (compacted row-major). bf16 elements (data_size code 1). Tensor dims are set
// equal to tile dims so no OOB clipping occurs. D# bit layout per ISA §8.3/8.4.
// This toolchain exposes the 6-arg builtin (g0, g1, g2, g3, g4, cpol).
// ---------------------------------------------------------------------------
static __device__ __forceinline__ void tdm_load_2d(unsigned lds_off,
                                                   const void* gaddr,
                                                   unsigned tile_w, unsigned tile_h,
                                                   unsigned long long stride0)
{
    unsigned long long ga = (unsigned long long)(size_t)gaddr;
    v4u g0;
    g0[0] = 1u;                                   // count=1, user mode, no gather
    g0[1] = lds_off;                              // lds_addr (bytes)
    g0[2] = (unsigned)(ga & 0xffffffffu);         // global_addr[31:0]
    g0[3] = (unsigned)((ga >> 32) & 0x1ffffffu)   // global_addr[56:32]
          | (2u << 30);                           // type = 2 ("image")
    v8i g1;
    g1[0] = (int)(1u << 16);                      // wg_mask=0, data_size=1 (2B)
    g1[1] = (int)((tile_w & 0xffffu) << 16);      // tensor_dim0[15:0]
    g1[2] = (int)((tile_w >> 16) | ((tile_h & 0xffffu) << 16)); // dim0 hi | dim1 lo
    g1[3] = (int)((tile_h >> 16) | ((tile_w & 0xffffu) << 16)); // dim1 hi | tile_dim0
    g1[4] = (int)(tile_h & 0xffffu);              // tile_dim1 (tile_dim2 = 0)
    g1[5] = (int)(stride0 & 0xffffffffu);         // tensor_dim0_stride[31:0]
    g1[6] = (int)((stride0 >> 32) & 0xffffu);     // stride hi (dim1_stride = 0)
    g1[7] = 0;
    v4i z4 = {};
    v8i z8 = {};
    __builtin_amdgcn_tensor_load_to_lds(g0, g1, z4, z4, z8, 0);
}

// ---------------------------------------------------------------------------
// Kernel 0: one-shot pre-pack of Wq/Wk/Wv into bf16 WMMA B-layout tiles:
//   Wp[mat][kstep][chunk][lane][16]  (32 B/lane, 288 KiB total -> stays in L2)
// ---------------------------------------------------------------------------
__global__ __launch_bounds__(256)
void pack_w_kernel(const float* __restrict__ Wq, const float* __restrict__ Wk,
                   const float* __restrict__ Wv, unsigned short* Wp_u)
{
    const int tid = blockIdx.x * 256 + threadIdx.x;
    if (tid >= 3 * KSTEPS * 4 * 32) return;
    const int lane = tid & 31;
    const int c    = (tid >> 5) & 3;
    const int ks   = (tid >> 7) % KSTEPS;
    const int mat  = (tid >> 7) / KSTEPS;

    const float* W = (mat == 0) ? Wq : (mat == 1) ? Wk : Wv;
    const int half = lane >> 4;
    const int l16  = lane & 15;

    v16bf bm;
    const float* wp = W + (long)(ks * 32 + half * 8) * HS + c * 16 + l16;
    #pragma unroll
    for (int v = 0; v < 8; ++v) {
        bm[2 * v]     = (__bf16)wp[(2 * v + (v >= 4 ? 8 : 0))     * HS];
        bm[2 * v + 1] = (__bf16)wp[(2 * v + 1 + (v >= 4 ? 8 : 0)) * HS];
    }
    *(v16bf*)((__bf16*)Wp_u + ((long)tid) * 16) = bm;
}

// ---------------------------------------------------------------------------
// Kernel 1: QKV projection via v_wmma_f32_16x16x32_bf16; each wave computes a
// 16x64 tile for one of q/k/v. Q pre-scaled by 1/sqrt(64); V stored transposed.
// ---------------------------------------------------------------------------
__global__ __launch_bounds__(256)
void qkv_proj_kernel(const float* __restrict__ x,
                     const float* __restrict__ bq, const float* __restrict__ bk,
                     const float* __restrict__ bv,
                     const unsigned short* __restrict__ Wp_u,
                     unsigned short* Qb_u, unsigned short* Kb_u, unsigned short* VTb_u)
{
    __bf16* Qb  = (__bf16*)Qb_u;
    __bf16* Kb  = (__bf16*)Kb_u;
    __bf16* VTb = (__bf16*)VTb_u;
    const __bf16* Wp = (const __bf16*)Wp_u;

    const int  lane = threadIdx.x & 31;
    const int  wave = threadIdx.x >> 5;
    const int  half = lane >> 4;
    const int  l16  = lane & 15;
    const int  mat  = blockIdx.y;            // 0=q, 1=k, 2=v
    const long rowBase = ((long)blockIdx.x * 8 + wave) * 16;

    const float* bias = (mat == 0) ? bq : (mat == 1) ? bk : bv;
    const __bf16* wbase = Wp + (((long)mat * KSTEPS * 4) * 32 + lane) * 16;

    v8f acc[4] = {};

    #pragma unroll 1
    for (int ks = 0; ks < KSTEPS; ++ks) {
        const int kk = ks * 32;
        const float* xrow = x + (rowBase + l16) * (long)EMB + kk + 8 * half;
        float4 x0 = *(const float4*)(xrow);
        float4 x1 = *(const float4*)(xrow + 4);
        float4 x2 = *(const float4*)(xrow + 16);
        float4 x3 = *(const float4*)(xrow + 20);
        v16bf a;
        a[0]=(__bf16)x0.x;  a[1]=(__bf16)x0.y;  a[2]=(__bf16)x0.z;  a[3]=(__bf16)x0.w;
        a[4]=(__bf16)x1.x;  a[5]=(__bf16)x1.y;  a[6]=(__bf16)x1.z;  a[7]=(__bf16)x1.w;
        a[8]=(__bf16)x2.x;  a[9]=(__bf16)x2.y;  a[10]=(__bf16)x2.z; a[11]=(__bf16)x2.w;
        a[12]=(__bf16)x3.x; a[13]=(__bf16)x3.y; a[14]=(__bf16)x3.z; a[15]=(__bf16)x3.w;

        #pragma unroll
        for (int c = 0; c < 4; ++c) {
            v16bf bm = *(const v16bf*)(wbase + ((long)(ks * 4 + c) * 32) * 16);
            acc[c] = __builtin_amdgcn_wmma_f32_16x16x32_bf16(
                false, a, false, bm, (short)0, acc[c], false, false);
        }
    }

    if (mat == 2) {
        #pragma unroll
        for (int c = 0; c < 4; ++c) {
            int   n  = c * 16 + l16;
            float bb = bias[n];
            v8bf pk;
            #pragma unroll
            for (int r = 0; r < 8; ++r) pk[r] = (__bf16)(acc[c][r] + bb);
            *(v8bf*)(VTb + (long)n * BT_TOTAL + rowBase + 8 * half) = pk;
        }
    } else {
        __bf16* outp = (mat == 0) ? Qb : Kb;
        float   sc   = (mat == 0) ? 0.125f : 1.0f;
        #pragma unroll
        for (int c = 0; c < 4; ++c) {
            int   n  = c * 16 + l16;
            float bb = bias[n];
            #pragma unroll
            for (int r = 0; r < 8; ++r) {
                long row = rowBase + r + 8 * half;
                outp[row * HS + n] = (__bf16)((acc[c][r] + bb) * sc);
            }
        }
    }
}

// ---------------------------------------------------------------------------
// Kernel 2: flash attention. All 8 waves of a block share each 32-key block,
// so wave 0 DMAs the K tile (32x64) and V^T tile (64x32) into LDS with the
// Tensor Data Mover, double-buffered (TENSORcnt), and every wave reads WMMA
// operands from LDS. Softmax is kept in registers; P re-layout via per-wave
// LDS staging.
// ---------------------------------------------------------------------------
__global__ __launch_bounds__(256)
void flash_attn_kernel(const unsigned short* Qb_u, const unsigned short* Kb_u,
                       const unsigned short* VTb_u, float* __restrict__ out)
{
    const __bf16* Qb  = (const __bf16*)Qb_u;
    const __bf16* Kb  = (const __bf16*)Kb_u;
    const __bf16* VTb = (const __bf16*)VTb_u;

    __shared__ __align__(32) __bf16 kbuf[2][32 * HS];   // K tiles   (2 x 4 KB)
    __shared__ __align__(32) __bf16 vbuf[2][HS * 32];   // V^T tiles (2 x 4 KB)
    __shared__ __align__(32) __bf16 pstage[8][16][32];  // per-wave P (8 KB)

    const int  lane = threadIdx.x & 31;
    const int  wave = threadIdx.x >> 5;
    const int  half = lane >> 4;
    const int  l16  = lane & 15;
    const long rowBase   = ((long)blockIdx.x * 8 + wave) * 16;
    const long batchBase = (rowBase / TSEQ) * TSEQ;   // keys stay in-batch

    // Q A-operands for the two K=32 steps over H=64
    v16bf qa[2];
    #pragma unroll
    for (int s = 0; s < 2; ++s) {
        const __bf16* qp = Qb + (rowBase + l16) * (long)HS + s * 32 + half * 16;
        v8bf lo = *(const v8bf*)qp;
        v8bf hi = *(const v8bf*)(qp + 8);
        qa[s] = join16(lo, hi);
    }

    // Kick off TDM for key block 0 (wave 0 only -> one DMA per tile)
    if (wave == 0) {
        tdm_load_2d((unsigned)(size_t)&kbuf[0][0], Kb + batchBase * HS, HS, 32, HS);
        tdm_load_2d((unsigned)(size_t)&vbuf[0][0], VTb + batchBase, 32, HS, BT_TOTAL);
    }

    v8f   O[4] = {};
    float m_i[8], l_i[8];
    #pragma unroll
    for (int r = 0; r < 8; ++r) { m_i[r] = -1e30f; l_i[r] = 0.0f; }

    #pragma unroll 1
    for (int j = 0; j < TSEQ; j += 32) {
        const int p = (j >> 5) & 1;
        if (wave == 0) {
            if (j + 32 < TSEQ) {   // issue next block, then wait current done
                const long nb = batchBase + j + 32;
                tdm_load_2d((unsigned)(size_t)&kbuf[p ^ 1][0], Kb + nb * HS, HS, 32, HS);
                tdm_load_2d((unsigned)(size_t)&vbuf[p ^ 1][0], VTb + nb, 32, HS, BT_TOTAL);
                __builtin_amdgcn_s_wait_tensorcnt(2);  // in-order: current 2 done
            } else {
                __builtin_amdgcn_s_wait_tensorcnt(0);
            }
        }
        __syncthreads();   // publish kbuf[p]/vbuf[p] to all waves

        // S = (Q*scale) K^T for two 16-key subblocks (operands from LDS)
        v8f st[2];
        #pragma unroll
        for (int t = 0; t < 2; ++t) {
            v8f c = {};
            #pragma unroll
            for (int s = 0; s < 2; ++s) {
                const __bf16* kp = &kbuf[p][(t * 16 + l16) * HS + s * 32 + half * 16];
                v16bf kb = *(const v16bf*)kp;
                c = __builtin_amdgcn_wmma_f32_16x16x32_bf16(
                    false, qa[s], false, kb, (short)0, c, false, false);
            }
            st[t] = c;
        }

        // Online softmax over this 32-key block
        float p0[8], p1[8], alpha[8];
        #pragma unroll
        for (int r = 0; r < 8; ++r) {
            float bm = fmaxf(st[0][r], st[1][r]);
            #pragma unroll
            for (int d = 1; d < 16; d <<= 1)
                bm = fmaxf(bm, __shfl_xor(bm, d, 32));
            float mn = fmaxf(m_i[r], bm);
            alpha[r] = __expf(m_i[r] - mn);
            float e0 = __expf(st[0][r] - mn);
            float e1 = __expf(st[1][r] - mn);
            float rs = e0 + e1;
            #pragma unroll
            for (int d = 1; d < 16; d <<= 1)
                rs += __shfl_xor(rs, d, 32);
            l_i[r] = l_i[r] * alpha[r] + rs;
            m_i[r] = mn;
            p0[r] = e0;
            p1[r] = e1;
        }
        #pragma unroll
        for (int c = 0; c < 4; ++c)
            #pragma unroll
            for (int r = 0; r < 8; ++r)
                O[c][r] *= alpha[r];

        // Stage P through per-wave LDS: C layout -> A layout
        #pragma unroll
        for (int r = 0; r < 8; ++r) {
            pstage[wave][8 * half + r][l16]      = (__bf16)p0[r];
            pstage[wave][8 * half + r][16 + l16] = (__bf16)p1[r];
        }
        asm volatile("s_wait_dscnt 0" ::: "memory");  // DS in-order per wave
        v8bf plo = *(const v8bf*)&pstage[wave][l16][half * 16];
        v8bf phi = *(const v8bf*)&pstage[wave][l16][half * 16 + 8];
        v16bf pa = join16(plo, phi);

        // O += P * V   (V^T tile rows contiguous over keys, from LDS)
        #pragma unroll
        for (int c = 0; c < 4; ++c) {
            const __bf16* vp = &vbuf[p][(c * 16 + l16) * 32 + half * 16];
            v16bf vb = *(const v16bf*)vp;
            O[c] = __builtin_amdgcn_wmma_f32_16x16x32_bf16(
                false, pa, false, vb, (short)0, O[c], false, false);
        }
        __syncthreads();   // all waves done with buffer p before reuse
    }

    // Normalize and store fp32 output [B,T,H]
    #pragma unroll
    for (int c = 0; c < 4; ++c) {
        int n = c * 16 + l16;
        #pragma unroll
        for (int r = 0; r < 8; ++r) {
            long row = rowBase + r + 8 * half;
            out[row * HS + n] = O[c][r] / l_i[r];
        }
    }
}

// ---------------------------------------------------------------------------
extern "C" void kernel_launch(void* const* d_in, const int* in_sizes, int n_in,
                              void* d_out, int out_size, void* d_ws, size_t ws_size,
                              hipStream_t stream) {
    const float* x  = (const float*)d_in[0];
    const float* Wq = (const float*)d_in[1];
    const float* bq = (const float*)d_in[2];
    const float* Wk = (const float*)d_in[3];
    const float* bk = (const float*)d_in[4];
    const float* Wv = (const float*)d_in[5];
    const float* bv = (const float*)d_in[6];
    float* out = (float*)d_out;

    // Workspace (bf16): Q [BT,64] | K [BT,64] | V^T [64,BT] | packed W tiles
    unsigned short* Qb  = (unsigned short*)d_ws;
    unsigned short* Kb  = Qb  + (size_t)BT_TOTAL * HS;
    unsigned short* VTb = Kb  + (size_t)BT_TOTAL * HS;
    unsigned short* Wp  = VTb + (size_t)BT_TOTAL * HS;   // 3*24*4*32*16 bf16

    pack_w_kernel<<<dim3((3 * KSTEPS * 4 * 32 + 255) / 256), 256, 0, stream>>>(
        Wq, Wk, Wv, Wp);

    dim3 gridP(BT_TOTAL / (16 * 8), 3, 1);
    qkv_proj_kernel<<<gridP, 256, 0, stream>>>(x, bq, bk, bv, Wp, Qb, Kb, VTb);

    dim3 gridA(BT_TOTAL / (16 * 8), 1, 1);
    flash_attn_kernel<<<gridA, 256, 0, stream>>>(Qb, Kb, VTb, out);
}
